// PointCNNLayer_64037962384012
// MI455X (gfx1250) — matrix operand order repaired and checked by reading, not compiled, not run on hardware
//
#include <hip/hip_runtime.h>
#include <hip/hip_bf16.h>

typedef _Float16 half8 __attribute__((ext_vector_type(8)));
typedef _Float16 v16h  __attribute__((ext_vector_type(16)));
typedef float    v8f   __attribute__((ext_vector_type(8)));

union V16H { v16h v; half8 h[2]; };

#define BATCH 2
#define NPTS  8192
#define RREP  2048
#define KNN   16
#define MROWS (BATCH * RREP)   // 4096 points total
#define CIN   64
#define CL    16
#define K2    256

// ---------------------------------------------------------------------------
// helpers
// ---------------------------------------------------------------------------
__device__ __forceinline__ half8 zero_h8() {
  half8 z;
#pragma unroll
  for (int i = 0; i < 8; ++i) z[i] = (_Float16)0.0f;
  return z;
}

// ---------------------------------------------------------------------------
// Kernel 0: one-time weight conversion f32 -> f16 with zero padding K -> Kpad.
// dst[n*Kpad + k] = k < K ? (f16)src[n*K + k] : 0
// ---------------------------------------------------------------------------
__global__ void __launch_bounds__(256) cvt_weights_kernel(
    const float* __restrict__ src, _Float16* __restrict__ dst,
    int N, int K, int Kpad) {
  const int idx = blockIdx.x * 256 + threadIdx.x;
  if (idx >= N * Kpad) return;
  const int n = idx / Kpad;
  const int k = idx - n * Kpad;
  dst[idx] = (k < K) ? (_Float16)src[(size_t)n * K + k] : (_Float16)0.0f;
}

// ---------------------------------------------------------------------------
// Kernel 1: KNN for representative points only (B*R blocks).
// ---------------------------------------------------------------------------
__global__ void __launch_bounds__(256) knn_kernel(
    const float* __restrict__ points,   // [B][N][3]
    const int*   __restrict__ rep_idx,  // [R]
    int*         __restrict__ nb_idx,   // [M][16]
    float*       __restrict__ p32,      // [M][48]
    _Float16*    __restrict__ pf16,     // [M][64]  (zero padded 48..63)
    float*       __restrict__ rep_out)  // [M][3]  (d_out part 1)
{
  __shared__ float dist[NPTS];
  __shared__ float smin[256];
  __shared__ int   simin[256];
  __shared__ int   sel[KNN + 1];

  const int tid = threadIdx.x;
  const int b   = blockIdx.x >> 11;       // / RREP
  const int r   = blockIdx.x & (RREP - 1);
  const int rid = rep_idx[r];

  const float rx = points[((size_t)b * NPTS + rid) * 3 + 0];
  const float ry = points[((size_t)b * NPTS + rid) * 3 + 1];
  const float rz = points[((size_t)b * NPTS + rid) * 3 + 2];
  const float rsq = rx * rx + ry * ry + rz * rz;

  for (int n = tid; n < NPTS; n += 256) {
    const float qx = points[((size_t)b * NPTS + n) * 3 + 0];
    const float qy = points[((size_t)b * NPTS + n) * 3 + 1];
    const float qz = points[((size_t)b * NPTS + n) * 3 + 2];
    const float qsq = qx * qx + qy * qy + qz * qz;
    dist[n] = rsq + qsq - 2.0f * (rx * qx + ry * qy + rz * qz);
  }
  __syncthreads();

  // 17 rounds of argmin (index tie-break = smaller index, matching top_k).
  for (int it = 0; it < KNN + 1; ++it) {
    float best = 3.4e38f;
    int   bidx = NPTS;
    for (int n = tid; n < NPTS; n += 256) {
      const float d = dist[n];
      if (d < best || (d == best && n < bidx)) { best = d; bidx = n; }
    }
    smin[tid] = best;
    simin[tid] = bidx;
    __syncthreads();
    for (int s = 128; s > 0; s >>= 1) {
      if (tid < s) {
        const float dv = smin[tid + s];
        const int   iv = simin[tid + s];
        if (dv < smin[tid] || (dv == smin[tid] && iv < simin[tid])) {
          smin[tid] = dv; simin[tid] = iv;
        }
      }
      __syncthreads();
    }
    if (tid == 0) {
      const int w = simin[0];
      sel[it] = w;
      dist[w] = 3.4e38f;
    }
    __syncthreads();
  }

  const size_t pt = (size_t)blockIdx.x;   // b*R + r
  if (tid < KNN) {
    const int nb = sel[tid + 1];          // drop self
    nb_idx[pt * KNN + tid] = nb;
    const float dx = points[((size_t)b * NPTS + nb) * 3 + 0] - rx;
    const float dy = points[((size_t)b * NPTS + nb) * 3 + 1] - ry;
    const float dz = points[((size_t)b * NPTS + nb) * 3 + 2] - rz;
    p32[pt * 48 + tid * 3 + 0] = dx;
    p32[pt * 48 + tid * 3 + 1] = dy;
    p32[pt * 48 + tid * 3 + 2] = dz;
    pf16[pt * 64 + tid * 3 + 0] = (_Float16)dx;
    pf16[pt * 64 + tid * 3 + 1] = (_Float16)dy;
    pf16[pt * 64 + tid * 3 + 2] = (_Float16)dz;
  } else if (tid >= 48 && tid < 64) {
    pf16[pt * 64 + tid] = (_Float16)0.0f;  // pad K 48 -> 64
  }
  if (tid == 0) {
    rep_out[pt * 3 + 0] = rx;
    rep_out[pt * 3 + 1] = ry;
    rep_out[pt * 3 + 2] = rz;
  }
}

// ---------------------------------------------------------------------------
// Kernel 2: lift (3->16 MLP + BN + ReLU) and neighbor-feature gather into
// transposed f16 layout featT[pt][c][j] (c = 0..79, j = 0..15).
// Block 0 also zeroes the 32-half zero-pad region passed to tf_kernel (which never reads it).
// ---------------------------------------------------------------------------
__global__ void __launch_bounds__(256) lift_gather_kernel(
    const float* __restrict__ features,  // [B][N][64]
    const int*   __restrict__ nb_idx,    // [M][16]
    const float* __restrict__ p32,       // [M][48]
    const float* __restrict__ lw,        // [16][3]
    const float* __restrict__ lb,        // [16]
    const float* __restrict__ lg,        // [16]
    const float* __restrict__ lbb,       // [16]
    _Float16*    __restrict__ featT,     // [M][80][16]
    _Float16*    __restrict__ zpad)      // [32] zeros
{
  const int tid = threadIdx.x;
  const size_t pt = (size_t)blockIdx.x;
  const int b = blockIdx.x >> 11;

  if (blockIdx.x == 0 && tid < 32) zpad[tid] = (_Float16)0.0f;

  __shared__ float sp[48];
  __shared__ int   snb[KNN];
  if (tid < 48) sp[tid] = p32[pt * 48 + tid];
  if (tid < KNN) snb[tid] = nb_idx[pt * KNN + tid];
  __syncthreads();

  {  // lift: 16 nbrs x 16 channels == 256 threads
    const int j = tid >> 4;
    const int c = tid & 15;
    float acc = lb[c];
#pragma unroll
    for (int d = 0; d < 3; ++d) acc += sp[j * 3 + d] * lw[c * 3 + d];
    float y = acc * lg[c] + lbb[c];
    y = fmaxf(y, 0.0f);
    featT[pt * 1280 + c * 16 + j] = (_Float16)y;
  }
  // neighbor features: 16 nbrs x 64 channels
  for (int idx = tid; idx < KNN * CIN; idx += 256) {
    const int cf = idx >> 4;
    const int j  = idx & 15;
    featT[pt * 1280 + (CL + cf) * 16 + j] =
        (_Float16)features[((size_t)b * NPTS + snb[j]) * CIN + cf];
  }
}

// ---------------------------------------------------------------------------
// Kernel 3: WMMA GEMM  O = act(bn(A @ W^T + bias)), all-f16 operands.
// A:  [M][KTILES*32] f16, zero padded
// Wh: [256][KTILES*32] f16, zero padded (pre-converted)
// O:  [M][256] f16.  8 waves, 2 N-tiles of 16 each; A fragments hoisted
// across both N-tiles; k-loop fully unrolled (KTILES is compile-time).
// Fragment layouts per CDNA5 ISA 7.12.2 (wave32).
// ---------------------------------------------------------------------------
template<int KTILES, int RELU>
__global__ void __launch_bounds__(256) gemm_bn_act_kernel(
    const _Float16* __restrict__ A,
    const _Float16* __restrict__ Wh,
    const float* __restrict__ bias,
    const float* __restrict__ g,
    const float* __restrict__ bb,
    _Float16* __restrict__ O)
{
  constexpr int LDK = KTILES * 32;
  const int lane = threadIdx.x & 31;
  const int wave = threadIdx.x >> 5;
  const int m0   = blockIdx.x * 16;
  const int row  = lane & 15;
  const int hsel = lane >> 4;            // 0: lanes 0-15, 1: lanes 16-31
  const int aofs = hsel * 8;             // A: K 0-7/16-23 vs 8-15/24-31
  const int bofs = hsel * 8;             // B: K 0-7/16-23 vs 8-15/24-31

  // hoist A fragments (reused by both N-tiles of this wave)
  V16H af[KTILES];
  {
    const _Float16* ap = A + (size_t)(m0 + row) * LDK + aofs;
#pragma unroll
    for (int kt = 0; kt < KTILES; ++kt) {
      af[kt].h[0] = *(const half8*)(ap + kt * 32);
      af[kt].h[1] = *(const half8*)(ap + kt * 32 + 16);
    }
  }

#pragma unroll
  for (int t = 0; t < 2; ++t) {
    const int col = (wave * 2 + t) * 16 + row;
    const _Float16* bp = Wh + (size_t)col * LDK + bofs;
    v8f acc = {};
#pragma unroll
    for (int kt = 0; kt < KTILES; ++kt) {
      V16H bfr;
      bfr.h[0] = *(const half8*)(bp + kt * 32);
      bfr.h[1] = *(const half8*)(bp + kt * 32 + 16);
      acc = __builtin_amdgcn_wmma_f32_16x16x32_f16(
          false, af[kt].v, false, bfr.v, (short)0, acc, false, false);
    }
    // epilogue: bias + batchnorm (+ relu), f16 store
    const float bi = bias[col];
    const float gg = g[col];
    const float gb = bb[col];
    const int rbase = hsel * 8;
#pragma unroll
    for (int rr = 0; rr < 8; ++rr) {
      float v = acc[rr] + bi;
      v = v * gg + gb;
      if constexpr (RELU) v = fmaxf(v, 0.0f);
      O[(size_t)(m0 + rbase + rr) * K2 + col] = (_Float16)v;
    }
  }
}

// ---------------------------------------------------------------------------
// Kernel 4: per-point transform Tf = T(16x16) @ feat(16x80), K padded 16->32.
// 8 waves per block, one point per wave, 5 N-tiles of 16 each.
// Branchless B fragments: elements 8-15 of every lane (K=16..31 of the padded
// K dimension) are zero registers, not loaded from any pad region,
// so all B-tile loads are unconditional and can be clause-issued ahead of
// the WMMA chain.
// ---------------------------------------------------------------------------
__global__ void __launch_bounds__(256) tf_kernel(
    const _Float16* __restrict__ T,      // [M][256] : T[pt][i*16+j]
    const _Float16* __restrict__ featT,  // [M][1280]: featT[pt][c*16+j]
    const _Float16* __restrict__ zpad,   // [32] zeros
    _Float16*       __restrict__ Tf)     // [M][1280]: Tf[pt][c*16+i]
{
  const int lane = threadIdx.x & 31;
  const int wave = threadIdx.x >> 5;
  const size_t pt = (size_t)blockIdx.x * 8 + wave;
  const int row  = lane & 15;
  const int hsel = lane >> 4;

  // A = T tile: lanes<16 supply K 0-7 (K 16-23 zero), lanes>=16 K 8-15.
  V16H a;
  a.h[0] = *(const half8*)(T + pt * 256 + row * 16 + hsel * 8);
  a.h[1] = zero_h8();

  const _Float16* fbase = featT + pt * 1280 + row * 16;
#pragma unroll
  for (int nt = 0; nt < 5; ++nt) {
    // B: lanes<16 hold K 0-7 (K 16-23 zero), lanes>=16 K 8-15 (K 24-31 zero).
    const _Float16* bp = fbase + nt * 256 + hsel * 8;
    V16H bfr;
    bfr.h[0] = *(const half8*)(bp);
    bfr.h[1] = zero_h8();
    v8f acc = {};
    acc = __builtin_amdgcn_wmma_f32_16x16x32_f16(
        false, a.v, false, bfr.v, (short)0, acc, false, false);
    half8 hv;
#pragma unroll
    for (int rr = 0; rr < 8; ++rr) hv[rr] = (_Float16)acc[rr];
    // store Tf[pt][c*16 + i], c = nt*16+row, i = rr + 8*hsel (contiguous 16B)
    *(half8*)(Tf + pt * 1280 + (size_t)(nt * 16 + row) * 16 + hsel * 8) = hv;
  }
}

// ---------------------------------------------------------------------------
// Kernel 5: final contraction out = Tf(4096x1280) @ conv_w^T + conv_b.
// conv_w[o][c][k] row-major == Wh[o][kc], kc = c*16+k (matches Tf layout);
// pre-converted to f16. N=128: 8 waves, one 16-col tile each. Streaming
// fragments (no A reuse within a wave). f32 output.
// ---------------------------------------------------------------------------
__global__ void __launch_bounds__(256) conv_gemm_kernel(
    const _Float16* __restrict__ Tf,   // [M][1280]
    const _Float16* __restrict__ CWh,  // [128][1280] f16
    const float*    __restrict__ cb,   // [128]
    float*          __restrict__ O)    // [M][128]
{
  constexpr int LDK = 1280;
  const int lane = threadIdx.x & 31;
  const int wave = threadIdx.x >> 5;
  const int m0   = blockIdx.x * 16;
  const int row  = lane & 15;
  const int hsel = lane >> 4;
  const int col  = wave * 16 + row;   // output channel

  const _Float16* ap = Tf + (size_t)(m0 + row) * LDK + hsel * 8;
  const _Float16* bp = CWh + (size_t)col * LDK + hsel * 8;

  v8f acc = {};
#pragma unroll
  for (int kt = 0; kt < LDK / 32; ++kt) {
    V16H a, bfr;
    a.h[0] = *(const half8*)(ap + kt * 32);
    a.h[1] = *(const half8*)(ap + kt * 32 + 16);
    bfr.h[0] = *(const half8*)(bp + kt * 32);
    bfr.h[1] = *(const half8*)(bp + kt * 32 + 16);
    acc = __builtin_amdgcn_wmma_f32_16x16x32_f16(
        false, a.v, false, bfr.v, (short)0, acc, false, false);
  }
  const float bi = cb[col];
  const int rbase = hsel * 8;
#pragma unroll
  for (int rr = 0; rr < 8; ++rr) {
    O[(size_t)(m0 + rbase + rr) * 128 + col] = acc[rr] + bi;
  }
}

// ---------------------------------------------------------------------------
// launch
// ---------------------------------------------------------------------------
extern "C" void kernel_launch(void* const* d_in, const int* in_sizes, int n_in,
                              void* d_out, int out_size, void* d_ws, size_t ws_size,
                              hipStream_t stream) {
  (void)in_sizes; (void)n_in; (void)out_size; (void)ws_size;

  const float* points    = (const float*)d_in[0];
  const float* features  = (const float*)d_in[1];
  const float* lift_w    = (const float*)d_in[2];
  const float* lift_b    = (const float*)d_in[3];
  const float* bn_lift_g = (const float*)d_in[4];
  const float* bn_lift_b = (const float*)d_in[5];
  const float* t1_w      = (const float*)d_in[6];
  const float* t1_b      = (const float*)d_in[7];
  const float* bn1_g     = (const float*)d_in[8];
  const float* bn1_b     = (const float*)d_in[9];
  const float* t2_w      = (const float*)d_in[10];
  const float* t2_b      = (const float*)d_in[11];
  const float* bn2_g     = (const float*)d_in[12];
  const float* bn2_b     = (const float*)d_in[13];
  const float* t3_w      = (const float*)d_in[14];
  const float* t3_b      = (const float*)d_in[15];
  const float* bn3_g     = (const float*)d_in[16];
  const float* bn3_b     = (const float*)d_in[17];
  const float* conv_w    = (const float*)d_in[18];
  const float* conv_b    = (const float*)d_in[19];
  const int*   rep_idx   = (const int*)d_in[20];

  float* out      = (float*)d_out;
  float* rep_out  = out;                       // [B][R][3] = 12288 floats
  float* conv_out = out + (size_t)MROWS * 3;   // [B][R][128]

  // workspace layout (bytes), all offsets 16B-aligned
  char* ws = (char*)d_ws;
  int*      nb    = (int*)(ws + 0);                 //   262144: [M][16] i32
  float*    p32   = (float*)(ws + 262144);          //   786432: [M][48] f32
  _Float16* pf16  = (_Float16*)(ws + 1048576);      //   524288: [M][64] f16
  _Float16* featT = (_Float16*)(ws + 1572864);      // 10485760: [M][1280] f16
  _Float16* h1    = (_Float16*)(ws + 12058624);     //  2097152: [M][256] f16
  _Float16* h2    = (_Float16*)(ws + 14155776);     //  2097152: [M][256] f16
  _Float16* Tm    = (_Float16*)(ws + 16252928);     //  2097152: [M][256] f16
  _Float16* Tf    = (_Float16*)(ws + 18350080);     // 10485760: [M][1280] f16
  _Float16* t1h   = (_Float16*)(ws + 28835840);     //    32768: [256][64]  f16
  _Float16* t2h   = (_Float16*)(ws + 28868608);     //   131072: [256][256] f16
  _Float16* t3h   = (_Float16*)(ws + 28999680);     //   131072: [256][256] f16
  _Float16* cwh   = (_Float16*)(ws + 29130752);     //   327680: [128][1280] f16
  _Float16* zpad  = (_Float16*)(ws + 29458432);     //       64: [32] f16 zeros
                                                    // total 29,458,496 B

  // weight pre-conversion (f32 -> f16, zero padded)
  cvt_weights_kernel<<<(256 * 64 + 255) / 256, 256, 0, stream>>>(t1_w, t1h, 256, 48, 64);
  cvt_weights_kernel<<<(256 * 256 + 255) / 256, 256, 0, stream>>>(t2_w, t2h, 256, 256, 256);
  cvt_weights_kernel<<<(256 * 256 + 255) / 256, 256, 0, stream>>>(t3_w, t3h, 256, 256, 256);
  cvt_weights_kernel<<<(128 * 1280 + 255) / 256, 256, 0, stream>>>(conv_w, cwh, 128, 1280, 1280);

  knn_kernel<<<MROWS, 256, 0, stream>>>(points, rep_idx, nb, p32, pf16, rep_out);
  lift_gather_kernel<<<MROWS, 256, 0, stream>>>(features, nb, p32, lift_w, lift_b,
                                                bn_lift_g, bn_lift_b, featT, zpad);
  // T-net: pf(48->64 pad) @ t1 -> h1 ; h1 @ t2 -> h2 ; h2 @ t3 -> T
  gemm_bn_act_kernel<2, 1><<<MROWS / 16, 256, 0, stream>>>(pf16, t1h, t1_b, bn1_g, bn1_b, h1);
  gemm_bn_act_kernel<8, 1><<<MROWS / 16, 256, 0, stream>>>(h1, t2h, t2_b, bn2_g, bn2_b, h2);
  gemm_bn_act_kernel<8, 0><<<MROWS / 16, 256, 0, stream>>>(h2, t3h, t3_b, bn3_g, bn3_b, Tm);
  // per-point Tf = T @ feat
  tf_kernel<<<MROWS / 8, 256, 0, stream>>>(Tm, featT, zpad, Tf);
  // out = Tf @ conv_w^T + conv_b
  conv_gemm_kernel<<<MROWS / 16, 256, 0, stream>>>(Tf, cwh, conv_b, conv_out);
}
